// Stem_8864812499419
// MI455X (gfx1250) — matrix operand order repaired and checked by reading, not compiled, not run on hardware
//
#include <hip/hip_runtime.h>

// ---------------------------------------------------------------------------
// CDNA5 (gfx1250) persistent tree-LSTM kernel.
// wave32, WMMA bf16 16x16x32, f32 accumulate.
// ---------------------------------------------------------------------------

typedef __bf16 bf16_t;
typedef bf16_t v16bf __attribute__((ext_vector_type(16)));
typedef float  v8f   __attribute__((ext_vector_type(8)));

#define B_TOT 256
#define SEQ_L 32
#define WDIM  300
#define WPAD  320          // K padded to multiple of 32
#define HID   64
#define G4    256          // 4*HID gate width
#define BT    16           // batch tile per workgroup (WMMA M)
#define NWG   (B_TOT/BT)   // 16 workgroups
#define KBX   (WPAD/32)    // 10 K-steps for input GEMM
#define KBH   (HID/32)     // 2 K-steps for hidden GEMM
#define NTN   (G4/16)      // 16 N-tiles of gates

#define SEG   528          // sum 1..32
#define ORI_BASE ((long)B_TOT*SEG*WDIM)
#define EXI_BASE (ORI_BASE + (long)B_TOT*SEG)

// ---- d_ws layout (bytes) ---------------------------------------------------
#define WIP_DIR      (KBX*NTN*32*16)          // 81920 bf16 elems per direction
#define WHP_DIR      (KBH*NTN*32*16)          // 16384 bf16 elems per direction
#define WS_WIPACK    0
#define WS_WHPACK    (2*WIP_DIR*2)            // 327680
#define WS_BIAS      (WS_WHPACK + 2*WHP_DIR*2)// 393216
#define WS_WGBASE    524288
#define UNIT_BYTES   (BT*SEQ_L*WPAD*4)        // 655360
#define GX_BYTES     (2*SEQ_L*NTN*256*4)      // 1048576  (fragment-native tiles)
#define HS_BYTES     (2*SEQ_L*BT*HID*4)       // 262144
#define WG_BYTES     (2*UNIT_BYTES + GX_BYTES + HS_BYTES)  // 2621440
// total ws need: 524288 + 16*2621440 = 42,467,328 B (~40.5 MB)

// ---------------------------------------------------------------------------
__device__ __forceinline__ unsigned short f2bfu(float f) {
  unsigned u = __builtin_bit_cast(unsigned, f);
  return (unsigned short)((u + 0x7FFFu + ((u >> 16) & 1u)) >> 16);
}
__device__ __forceinline__ bf16_t f2bf(float f) {
  unsigned short s = f2bfu(f);
  return __builtin_bit_cast(bf16_t, s);
}
__device__ __forceinline__ float sigm(float x) {
  return 1.0f / (1.0f + __expf(-x));
}
// Build one 16x32 bf16 A-fragment from two contiguous 8-float runs
// (k = kb*32 + half*16 + {0..7}, k = kb*32 + half*16 + 8 + {0..7}).
__device__ __forceinline__ v16bf make_afrag(const float4* rp, int kb, int half) {
  int q0 = kb * 8 + half * 4;        // float4 index of first run
  int q1 = q0 + 2;                   // float4 index of second run
  float4 a0 = rp[q0], a1 = rp[q0 + 1], a2 = rp[q1], a3 = rp[q1 + 1];
  v16bf a;
  a[0] = f2bf(a0.x);  a[1] = f2bf(a0.y);  a[2]  = f2bf(a0.z);  a[3]  = f2bf(a0.w);
  a[4] = f2bf(a1.x);  a[5] = f2bf(a1.y);  a[6]  = f2bf(a1.z);  a[7]  = f2bf(a1.w);
  a[8] = f2bf(a2.x);  a[9] = f2bf(a2.y);  a[10] = f2bf(a2.z);  a[11] = f2bf(a2.w);
  a[12] = f2bf(a3.x); a[13] = f2bf(a3.y); a[14] = f2bf(a3.z);  a[15] = f2bf(a3.w);
  return a;
}

// ---------------------------------------------------------------------------
// Prep: pack Wi^T / Wh^T (bf16) into WMMA-B-fragment-native layout:
//   pack[((kb*16 + nt)*32 + lane)*16 + j] = W[n][k],
//   n = nt*16 + (lane&15), k = kb*32 + (lane>>4)*16 + j   (zero-padded k>=K)
// so a lane's B fragment is one contiguous 32B load.
// Also biasSum[dir][n] = bi[n] + bh[n].
// ---------------------------------------------------------------------------
__global__ void prep_pack(const float* __restrict__ Wi_f, const float* __restrict__ Wh_f,
                          const float* __restrict__ bi_f, const float* __restrict__ bh_f,
                          const float* __restrict__ Wi_b, const float* __restrict__ Wh_b,
                          const float* __restrict__ bi_b, const float* __restrict__ bh_b,
                          unsigned short* __restrict__ wiPack,
                          unsigned short* __restrict__ whPack,
                          float* __restrict__ biasSum) {
  int tid = blockIdx.x * blockDim.x + threadIdx.x;
  int stride = gridDim.x * blockDim.x;

  for (int i = tid; i < 2 * WIP_DIR; i += stride) {
    int dir = i / WIP_DIR;  int r = i % WIP_DIR;
    int j  = r & 15;        int lane = (r >> 4) & 31;
    int nt = (r >> 9) & 15; int kb   = r >> 13;           // kb < 10
    int n = nt * 16 + (lane & 15);
    int k = kb * 32 + ((lane >> 4) << 4) + j;
    const float* Wi = dir ? Wi_b : Wi_f;
    float v = (k < WDIM) ? Wi[n * WDIM + k] : 0.0f;
    wiPack[i] = f2bfu(v);
  }
  for (int i = tid; i < 2 * WHP_DIR; i += stride) {
    int dir = i / WHP_DIR;  int r = i % WHP_DIR;
    int j  = r & 15;        int lane = (r >> 4) & 31;
    int nt = (r >> 9) & 15; int kb   = r >> 13;           // kb < 2
    int n = nt * 16 + (lane & 15);
    int k = kb * 32 + ((lane >> 4) << 4) + j;             // k < 64
    const float* Wh = dir ? Wh_b : Wh_f;
    whPack[i] = f2bfu(Wh[n * HID + k]);
  }
  for (int i = tid; i < 2 * G4; i += stride) {
    int dir = i >> 8; int n = i & 255;
    biasSum[i] = dir ? (bi_b[n] + bh_b[n]) : (bi_f[n] + bh_f[n]);
  }
}

// ---------------------------------------------------------------------------
// Init: per-WG unit buffers [m][pos][WPAD] f32: unit = unit_hidden * exist,
// K-pad zeroed in both ping-pong buffers (combine only touches d<300).
// ---------------------------------------------------------------------------
__global__ void init_units(const float* __restrict__ unit_hidden,
                           const float* __restrict__ existence,
                           char* __restrict__ wgws) {
  int tid = blockIdx.x * blockDim.x + threadIdx.x;
  int stride = gridDim.x * blockDim.x;
  const int total = NWG * BT * SEQ_L * WPAD;
  for (int i = tid; i < total; i += stride) {
    int d = i % WPAD;  int r = i / WPAD;
    int p = r % SEQ_L; r /= SEQ_L;
    int m = r % BT;    int wg = r / BT;
    int b = wg * BT + m;
    float e = existence[b * SEQ_L + p];
    float v = (d < WDIM) ? unit_hidden[(b * SEQ_L + p) * WDIM + d] * e : 0.0f;
    float* uA = (float*)(wgws + (size_t)wg * WG_BYTES);
    float* uB = uA + UNIT_BYTES / 4;
    uA[(m * SEQ_L + p) * WPAD + d] = v;
    uB[(m * SEQ_L + p) * WPAD + d] = 0.0f;
  }
}

// ---------------------------------------------------------------------------
// Main persistent kernel: 16 WGs x 512 threads (16 waves). One WG = 16 batch
// rows through all 32 length-iterations.
// ---------------------------------------------------------------------------
__global__ __launch_bounds__(512, 1)
void tree_lstm_main(const float* __restrict__ existence,
                    const float* __restrict__ Wo, const float* __restrict__ bo,
                    const float* __restrict__ h0, const float* __restrict__ c0,
                    const unsigned short* __restrict__ wiPack,
                    const unsigned short* __restrict__ whPack,
                    const float* __restrict__ biasSum,
                    char* __restrict__ wgws0,
                    float* __restrict__ out) {
  __shared__ unsigned short sWh[2 * WHP_DIR] __attribute__((aligned(32))); // 64KB
  __shared__ float sH[2][2][BT][HID];   // [step-parity][dir][m][k]  16KB
  __shared__ float sC[2][BT][HID];      // 8KB
  __shared__ float sBias[2 * G4];
  __shared__ float sWo[2 * HID];
  __shared__ float sExA[BT][SEQ_L + 1];
  __shared__ float sExB[BT][SEQ_L + 1];
  __shared__ float sOr[BT][SEQ_L];

  const int wg   = blockIdx.x;
  const int tid  = threadIdx.x;
  const int wave = tid >> 5;
  const int lane = tid & 31;
  const int half = lane >> 4;
  const int lm   = lane & 15;
  const int rdir = (wave >> 2) & 1;   // recurrence direction for waves 0-7
  const int w4   = wave & 3;          // hidden-column block [w4*16, w4*16+16)

  char* base = wgws0 + (size_t)wg * WG_BYTES;
  float* uA = (float*)base;
  float* uB = (float*)(base + UNIT_BYTES);
  float* Gx = (float*)(base + 2 * UNIT_BYTES);           // tiles: [dir][t][nt][lane][8]
  float* Hs = (float*)(base + 2 * UNIT_BYTES + GX_BYTES);// [dir][t][m][64] f32

  // --- one-time LDS fills ---
  for (int i = tid; i < 2 * WHP_DIR; i += 512) sWh[i] = whPack[i];
  for (int i = tid; i < 2 * G4;      i += 512) sBias[i] = biasSum[i];
  for (int i = tid; i < 2 * HID;     i += 512) sWo[i] = Wo[i];
  for (int i = tid; i < BT * SEQ_L;  i += 512) {
    int m = i / SEQ_L, p = i % SEQ_L;
    sExA[m][p] = existence[(wg * BT + m) * SEQ_L + p];
  }
  __syncthreads();

  // per-wave kernel-constant gate biases (dir/w4/lm fixed per wave)
  float bsum[4];
#pragma unroll
  for (int g = 0; g < 4; ++g)
    bsum[g] = sBias[rdir * G4 + (g * 4 + w4) * 16 + lm];

  int cur = 0;
  for (int l = SEQ_L; l >= 1; --l) {
    float* uCur   = cur ? uB : uA;
    float* uNext  = cur ? uA : uB;
    float* exCur  = cur ? &sExB[0][0] : &sExA[0][0];  // stride SEQ_L+1
    float* exNext = cur ? &sExA[0][0] : &sExB[0][0];

    // reset h,c to h0/c0 for this length (h into parity-0 buffer)
    for (int i = tid; i < 2 * BT * HID; i += 512) {
      int k = i % HID; int m = (i / HID) % BT; int dir = i / (HID * BT);
      sH[0][dir][m][k] = h0[dir * HID + k];
      sC[dir][m][k]    = c0[dir * HID + k];
    }
    __syncthreads();

    // ---- Phase A: input projections Gx[dir][t] = X_t @ Wi^T  (bf16 WMMA) ---
    for (int job = wave; job < 2 * l; job += 16) {
      int adir = job & 1, t = job >> 1;
      const float4* xrp = (const float4*)(uCur + (lm * SEQ_L + t) * WPAD);
      v16bf afr[KBX];
#pragma unroll
      for (int kb = 0; kb < KBX; ++kb) afr[kb] = make_afrag(xrp, kb, half);
      const unsigned short* wp = wiPack + (size_t)adir * WIP_DIR;
      float* gx = Gx + (size_t)(adir * SEQ_L + t) * NTN * 256;
      for (int nt = 0; nt < NTN; ++nt) {
        v8f acc = {};
#pragma unroll
        for (int kb = 0; kb < KBX; ++kb) {
          v16bf bfr = *(const v16bf*)(wp + ((kb * NTN + nt) * 32 + lane) * 16);
          acc = __builtin_amdgcn_wmma_f32_16x16x32_bf16(
              false, afr[kb], false, bfr, (short)0, acc, false, false);
        }
        *(v8f*)(gx + nt * 256 + lane * 8) = acc;   // fragment-native 32B store
      }
    }
    __syncthreads();

    // ---- Phase B: recurrence. waves 0-3 = fwd, waves 4-7 = bwd ------------
    // Hoist the wave's 8 Wh B-fragments (constant across steps).
    v16bf whfr[4][KBH];
    if (wave < 8) {
#pragma unroll
      for (int g = 0; g < 4; ++g)
#pragma unroll
        for (int kb = 0; kb < KBH; ++kb)
          whfr[g][kb] = *(const v16bf*)(sWh + rdir * WHP_DIR +
                          ((kb * NTN + (g * 4 + w4)) * 32 + lane) * 16);
    }
    for (int s = 0; s < l; ++s) {
      int pb = s & 1;
      if (wave < 8) {
        int t = rdir ? (l - 1 - s) : s;
        const float4* hp = (const float4*)&sH[pb][rdir][lm][0];
        v16bf ah[KBH];
#pragma unroll
        for (int kb = 0; kb < KBH; ++kb) ah[kb] = make_afrag(hp, kb, half);

        const float* gxb = Gx + (size_t)(rdir * SEQ_L + t) * NTN * 256;
        float gate[4][8];
#pragma unroll
        for (int g = 0; g < 4; ++g) {
          v8f acc = {};
#pragma unroll
          for (int kb = 0; kb < KBH; ++kb)
            acc = __builtin_amdgcn_wmma_f32_16x16x32_bf16(
                false, ah[kb], false, whfr[g][kb], (short)0, acc, false, false);
          v8f gxv = *(const v8f*)(gxb + (g * 4 + w4) * 256 + lane * 8);
#pragma unroll
          for (int r = 0; r < 8; ++r)
            gate[g][r] = acc[r] + gxv[r] + bsum[g];
        }
        float* hsd = Hs + (size_t)(rdir * SEQ_L + t) * BT * HID;
#pragma unroll
        for (int r = 0; r < 8; ++r) {
          int m  = r + half * 8;
          int hc = w4 * 16 + lm;
          float i_ = sigm(gate[0][r]);
          float f_ = sigm(gate[1][r]);
          float g_ = tanhf(gate[2][r]);
          float o_ = sigm(gate[3][r]);
          float cn = f_ * sC[rdir][m][hc] + i_ * g_;
          float hn = o_ * tanhf(cn);
          sC[rdir][m][hc]       = cn;
          sH[1 - pb][rdir][m][hc] = hn;   // write next-parity buffer (no RAW race)
          hsd[m * HID + hc]     = hn;
        }
      }
      __syncthreads();  // single barrier per step: h_t visible for step s+1
    }

    // ---- Phase C: orient = [hf|hb] @ Wo^T + bo; emit unit/orient/exist ----
    const long offl = (long)l * (l - 1) / 2;
    for (int idx = tid; idx < BT * l; idx += 512) {
      int t = idx / BT, m = idx % BT;
      const float4* hf = (const float4*)(Hs + (size_t)(0 * SEQ_L + t) * BT * HID + m * HID);
      const float4* hb = (const float4*)(Hs + (size_t)(1 * SEQ_L + t) * BT * HID + m * HID);
      const float4* wf = (const float4*)&sWo[0];
      const float4* wb = (const float4*)&sWo[HID];
      float o = bo[0];
#pragma unroll
      for (int q = 0; q < HID / 4; ++q) {
        float4 a = hf[q], b = hb[q], x = wf[q], y = wb[q];
        o += a.x * x.x + a.y * x.y + a.z * x.z + a.w * x.w;
        o += b.x * y.x + b.y * y.y + b.z * y.z + b.w * y.w;
      }
      sOr[m][t] = o;
      int b = wg * BT + m;
      long pos = offl + t;
      out[ORI_BASE + (long)b * SEG + pos] = o;
      out[EXI_BASE + (long)b * SEG + pos] = exCur[m * (SEQ_L + 1) + t];
    }
    for (int i = tid; i < BT * l * WDIM; i += 512) {
      int d = i % WDIM;
      int t = (i / WDIM) % l;
      int m = i / (WDIM * l);
      int b = wg * BT + m;
      out[((long)b * SEG + offl + t) * WDIM + d] = uCur[(m * SEQ_L + t) * WPAD + d];
    }
    __syncthreads();  // sOr complete before combine

    if (l == 1) break;

    // ---- Phase D: combine (l -> l-1), ping-pong exist & unit --------------
    for (int idx = tid; idx < BT * (l - 1); idx += 512) {
      int p = idx / BT, m = idx % BT;
      float r0 = sOr[m][p]     > 0.0f ? 1.0f : 0.0f;
      float r1 = sOr[m][p + 1] > 0.0f ? 1.0f : 0.0f;
      float lw = r0 * exCur[m * (SEQ_L + 1) + p];
      float rw = (1.0f - r1) * exCur[m * (SEQ_L + 1) + p + 1];
      exNext[m * (SEQ_L + 1) + p] = fmaxf(lw, rw);
    }
    for (int i = tid; i < BT * (l - 1) * WDIM; i += 512) {
      int d = i % WDIM;
      int p = (i / WDIM) % (l - 1);
      int m = i / (WDIM * (l - 1));
      float r0 = sOr[m][p]     > 0.0f ? 1.0f : 0.0f;
      float r1 = sOr[m][p + 1] > 0.0f ? 1.0f : 0.0f;
      float lw = r0 * exCur[m * (SEQ_L + 1) + p];
      float rw = (1.0f - r1) * exCur[m * (SEQ_L + 1) + p + 1];
      uNext[(m * SEQ_L + p) * WPAD + d] =
          lw * uCur[(m * SEQ_L + p) * WPAD + d] +
          rw * uCur[(m * SEQ_L + p + 1) * WPAD + d];
    }
    __syncthreads();
    cur ^= 1;
  }
}

// ---------------------------------------------------------------------------
extern "C" void kernel_launch(void* const* d_in, const int* in_sizes, int n_in,
                              void* d_out, int out_size, void* d_ws, size_t ws_size,
                              hipStream_t stream) {
  const float* unit_hidden = (const float*)d_in[0];
  const float* existence   = (const float*)d_in[1];
  const float* Wi_f = (const float*)d_in[2];
  const float* Wh_f = (const float*)d_in[3];
  const float* bi_f = (const float*)d_in[4];
  const float* bh_f = (const float*)d_in[5];
  const float* Wi_b = (const float*)d_in[6];
  const float* Wh_b = (const float*)d_in[7];
  const float* bi_b = (const float*)d_in[8];
  const float* bh_b = (const float*)d_in[9];
  const float* Wo   = (const float*)d_in[10];
  const float* bo   = (const float*)d_in[11];
  const float* h0   = (const float*)d_in[12];
  const float* c0   = (const float*)d_in[13];

  char* ws = (char*)d_ws;
  unsigned short* wiPack = (unsigned short*)(ws + WS_WIPACK);
  unsigned short* whPack = (unsigned short*)(ws + WS_WHPACK);
  float*          biasSum = (float*)(ws + WS_BIAS);
  char*           wgws = ws + WS_WGBASE;

  prep_pack<<<64, 256, 0, stream>>>(Wi_f, Wh_f, bi_f, bh_f,
                                    Wi_b, Wh_b, bi_b, bh_b,
                                    wiPack, whPack, biasSum);
  init_units<<<2048, 256, 0, stream>>>(unit_hidden, existence, wgws);
  tree_lstm_main<<<NWG, 512, 0, stream>>>(existence, Wo, bo, h0, c0,
                                          wiPack, whPack, biasSum,
                                          wgws, (float*)d_out);
}
